// LNO_51075751084699
// MI455X (gfx1250) — compile-verified, matrix operand order checked by reading, not run on hardware
//
#include <hip/hip_runtime.h>
#include <hip/hip_bf16.h>

// ---------------------------------------------------------------------------
// LNO forward for MI455X (gfx1250, wave32, WMMA f32<-f16 16x16x32)
// ---------------------------------------------------------------------------

typedef __attribute__((ext_vector_type(16))) _Float16 v16h;
typedef __attribute__((ext_vector_type(8)))  _Float16 v8h;
typedef __attribute__((ext_vector_type(8)))  float    v8f;

#define S_LEN  2048
#define WCH    64
#define MMODE  64
#define BATCH  32
#define PI_F   3.14159265358979323846f

// --- WMMA wrapper (codegen-verified builtin, 8-arg form) -------------------
__device__ __forceinline__ v8f wmma_f16(v16h a, v16h b, v8f c) {
  return __builtin_amdgcn_wmma_f32_16x16x32_f16(
      /*neg_a=*/false, a, /*neg_b=*/false, b,
      /*c_mod=*/(short)0, c, /*reuse_a=*/false, /*reuse_b=*/false);
}

// A/B fragment from a K-contiguous f16 row (global or LDS).
// Per ISA 7.12.2 (16-bit A 16x32): lanes 0-15 hold K {k0+0..7, k0+16..23},
// lanes 16-31 hold K {k0+8..15, k0+24..31}; element pairs pack into VGPRs.
__device__ __forceinline__ v16h frag_h16(const _Float16* row, int k0, int lane) {
  int half = (lane >> 4) & 1;
  v8h lo = *reinterpret_cast<const v8h*>(row + k0 + 8 * half);
  v8h hi = *reinterpret_cast<const v8h*>(row + k0 + 16 + 8 * half);
  v16h f;
#pragma unroll
  for (int i = 0; i < 8; ++i) { f[i] = lo[i]; f[8 + i] = hi[i]; }
  return f;
}

// Same fragment sourced from a K-contiguous f32 row (weights), cvt to f16.
__device__ __forceinline__ v16h frag_f32(const float* row, int k0, int lane) {
  int half = (lane >> 4) & 1;
  const float* q = row + k0 + 8 * half;
  v16h f;
#pragma unroll
  for (int i = 0; i < 8; ++i) { f[i] = (_Float16)q[i]; f[8 + i] = (_Float16)q[16 + i]; }
  return f;
}

__device__ __forceinline__ float gelu_exact(float x) {
  return 0.5f * x * (1.0f + erff(x * 0.70710678118654752f));
}

// ---------------------------------------------------------------------------
// 1. Analytic Laplace eigenbasis: E[m,s] = sqrt(2/(S+1)) sin((m+1)(S-s)pi/(S+1))
//    Stored both ways: Gms [M][S] (proj B-frags), Gsm [S][M] (recon B-frags).
// ---------------------------------------------------------------------------
__global__ void lno_basis_kernel(_Float16* __restrict__ Gms,
                                 _Float16* __restrict__ Gsm) {
  int idx = blockIdx.x * 256 + threadIdx.x;   // 64*2048 threads
  int m = idx >> 11;
  int s = idx & (S_LEN - 1);
  const float c = sqrtf(2.0f / (float)(S_LEN + 1));
  float v = c * sinf((float)(m + 1) * (float)(S_LEN - s) * (PI_F / (float)(S_LEN + 1)));
  Gms[m * S_LEN + s] = (_Float16)v;
  Gsm[s * MMODE + m] = (_Float16)v;
}

// ---------------------------------------------------------------------------
// 2. Encoder: h[b,w,s] = sum_c x[b,c,s] * enc_w[c,w] + enc_b[w]   (Cin=8)
// ---------------------------------------------------------------------------
__global__ void lno_encoder_kernel(const float* __restrict__ x,
                                   const float* __restrict__ ew,
                                   const float* __restrict__ eb,
                                   float* __restrict__ h) {
  int idx = blockIdx.x * 256 + threadIdx.x;   // (b*64+w)*2048+s
  int s = idx & (S_LEN - 1);
  int r = idx >> 11;
  int b = r >> 6, w = r & 63;
  float acc = eb[w];
#pragma unroll
  for (int c = 0; c < 8; ++c)
    acc += x[((b * 8 + c) << 11) + s] * ew[c * WCH + w];
  h[idx] = acc;
}

// ---------------------------------------------------------------------------
// 3. InstanceNorm over S per row, emit f16 (WMMA A-operand layout, K-contig).
// ---------------------------------------------------------------------------
__global__ void lno_inorm_kernel(const float* __restrict__ in,
                                 _Float16* __restrict__ out) {
  int r = blockIdx.x;
  const float* row = in + (size_t)r * S_LEN;
  float s = 0.f, s2 = 0.f, v[8];
#pragma unroll
  for (int i = 0; i < 8; ++i) {
    float t = row[threadIdx.x + 256 * i];
    v[i] = t; s += t; s2 += t * t;
  }
  __shared__ float rs[256], rq[256];
  rs[threadIdx.x] = s; rq[threadIdx.x] = s2;
  __syncthreads();
  for (int o = 128; o > 0; o >>= 1) {
    if (threadIdx.x < o) { rs[threadIdx.x] += rs[threadIdx.x + o]; rq[threadIdx.x] += rq[threadIdx.x + o]; }
    __syncthreads();
  }
  float mean = rs[0] * (1.0f / S_LEN);
  float var  = rq[0] * (1.0f / S_LEN) - mean * mean;
  float inv  = rsqrtf(var + 1e-5f);
#pragma unroll
  for (int i = 0; i < 8; ++i)
    out[(size_t)r * S_LEN + threadIdx.x + 256 * i] = (_Float16)((v[i] - mean) * inv);
}

// ---------------------------------------------------------------------------
// 4. Projection GEMM: xs[r,m] = sum_s hn[r,s] * Gms[m,s]   (2048x64, K=2048)
//    1 wave = one 16x16 tile, 64 WMMAs down K.
// ---------------------------------------------------------------------------
__global__ void lno_proj_gemm_kernel(const _Float16* __restrict__ A,    // hn [2048][2048]
                                     const _Float16* __restrict__ Gms,  // [64][2048]
                                     float* __restrict__ xs) {          // [2048][64]
  int lane = threadIdx.x & 31, wv = threadIdx.x >> 5;
  int tile = blockIdx.x * 8 + wv;          // 512 tiles
  int mt = tile >> 2, nt = tile & 3;
  const _Float16* arp = A   + (size_t)(mt * 16 + (lane & 15)) * S_LEN;
  const _Float16* brp = Gms + (size_t)(nt * 16 + (lane & 15)) * S_LEN;
  v8f acc = {};
  for (int k0 = 0; k0 < S_LEN; k0 += 32) {
    __builtin_prefetch(arp + k0 + 256, 0, 1);   // global_prefetch_b8
    v16h a = frag_h16(arp, k0, lane);
    v16h b = frag_h16(brp, k0, lane);
    acc = wmma_f16(a, b, acc);
  }
  int half = lane >> 4, nc = lane & 15;
#pragma unroll
  for (int rr = 0; rr < 8; ++rr)
    xs[(size_t)(mt * 16 + rr + 8 * half) * MMODE + nt * 16 + nc] = acc[rr];
}

// ---------------------------------------------------------------------------
// 5. Per-mode channel mix: spec[b,o,m] = sum_i xs[b,i,m] * lw[i,o,m]  (tiny)
// ---------------------------------------------------------------------------
__global__ void lno_modemix_kernel(const float* __restrict__ xs,
                                   const float* __restrict__ lw,   // [64][64][64] i,o,m
                                   _Float16* __restrict__ spec) {  // [2048][64]
  int idx = blockIdx.x * 256 + threadIdx.x;   // (b*64+o)*64+m
  int m = idx & 63, o = (idx >> 6) & 63, b = idx >> 12;
  float acc = 0.f;
  for (int i = 0; i < WCH; ++i)
    acc += xs[((b << 6) + i) * MMODE + m] * lw[(i * WCH + o) * MMODE + m];
  spec[idx] = (_Float16)acc;
}

// ---------------------------------------------------------------------------
// 6. Reconstruction GEMM: y[r,s] = sum_m spec[r,m] * Gsm[s,m]   (K=64)
// ---------------------------------------------------------------------------
__global__ void lno_recon_gemm_kernel(const _Float16* __restrict__ spec, // [2048][64]
                                      const _Float16* __restrict__ Gsm,  // [2048][64]
                                      float* __restrict__ y) {           // [2048][2048]
  int lane = threadIdx.x & 31, wv = threadIdx.x >> 5;
  int tile = blockIdx.x * 8 + wv;          // 16384 tiles
  int mt = tile >> 7, nt = tile & 127;
  const _Float16* arp = spec + (size_t)(mt * 16 + (lane & 15)) * MMODE;
  const _Float16* brp = Gsm  + (size_t)(nt * 16 + (lane & 15)) * MMODE;
  v8f acc = {};
  acc = wmma_f16(frag_h16(arp, 0, lane),  frag_h16(brp, 0, lane),  acc);
  acc = wmma_f16(frag_h16(arp, 32, lane), frag_h16(brp, 32, lane), acc);
  int half = lane >> 4, nc = lane & 15;
#pragma unroll
  for (int rr = 0; rr < 8; ++rr)
    y[(size_t)(mt * 16 + rr + 8 * half) * S_LEN + nt * 16 + nc] = acc[rr];
}

// ---------------------------------------------------------------------------
// 7. Fused per-block tail: h = gelu( W2@gelu(W1@yn + b1) + b2 + Ww@h + wb )
//    WG = (batch b, 64-position s-tile). Activations staged transposed in LDS
//    so B-fragments are contiguous ds_load_b128; A = f32 weight rows.
// ---------------------------------------------------------------------------
#define TPAD 72   // LDS row stride (f16), keeps rows 16B aligned, breaks bank cycles

__global__ void lno_ffn_kernel(const _Float16* __restrict__ yn,
                               float* __restrict__ h,
                               const float* __restrict__ w1, const float* __restrict__ b1,
                               const float* __restrict__ w2, const float* __restrict__ b2,
                               const float* __restrict__ ww, const float* __restrict__ wb) {
  int b  = blockIdx.x >> 5;
  int s0 = (blockIdx.x & 31) << 6;
  __shared__ _Float16 TY[64 * TPAD];   // [pos][ch] normalized spectral branch
  __shared__ _Float16 TH[64 * TPAD];   // [pos][ch] skip branch (h)
  __shared__ _Float16 T1[64 * TPAD];   // [pos][ch] hidden
  int t = threadIdx.x;
  int ch = t >> 2, pos0 = (t & 3) << 4;
  {
    const _Float16* sy = yn + ((size_t)(b * WCH + ch) << 11) + s0 + pos0;
    const float*    sh = h  + ((size_t)(b * WCH + ch) << 11) + s0 + pos0;
#pragma unroll
    for (int i = 0; i < 16; ++i) {
      TY[(pos0 + i) * TPAD + ch] = sy[i];
      TH[(pos0 + i) * TPAD + ch] = (_Float16)sh[i];
    }
  }
  __syncthreads();

  int lane = t & 31, wv = t >> 5;
  int half = lane >> 4, lm = lane & 15;

  // Stage 1: T1 = gelu(W1 @ TY + b1). 16 tiles, 2 per wave.
#pragma unroll
  for (int tt = 0; tt < 2; ++tt) {
    int id = wv * 2 + tt, mt = id >> 2, nt = id & 3;
    const float*    ar = w1 + (mt * 16 + lm) * WCH;
    const _Float16* br = &TY[(nt * 16 + lm) * TPAD];
    v8f acc = {};
    acc = wmma_f16(frag_f32(ar, 0, lane),  frag_h16(br, 0, lane),  acc);
    acc = wmma_f16(frag_f32(ar, 32, lane), frag_h16(br, 32, lane), acc);
#pragma unroll
    for (int rr = 0; rr < 8; ++rr) {
      int o = mt * 16 + rr + 8 * half, p = nt * 16 + lm;
      T1[p * TPAD + o] = (_Float16)gelu_exact(acc[rr] + b1[o]);
    }
  }
  __syncthreads();

  // Stage 2+3 fused: out = gelu( W2@T1 + b2 + Ww@TH + wb ), written in place.
#pragma unroll
  for (int tt = 0; tt < 2; ++tt) {
    int id = wv * 2 + tt, mt = id >> 2, nt = id & 3;
    const float*    a2 = w2 + (mt * 16 + lm) * WCH;
    const float*    a3 = ww + (mt * 16 + lm) * WCH;
    const _Float16* bh = &T1[(nt * 16 + lm) * TPAD];
    const _Float16* bs = &TH[(nt * 16 + lm) * TPAD];
    v8f acc = {};
    acc = wmma_f16(frag_f32(a2, 0, lane),  frag_h16(bh, 0, lane),  acc);
    acc = wmma_f16(frag_f32(a2, 32, lane), frag_h16(bh, 32, lane), acc);
    acc = wmma_f16(frag_f32(a3, 0, lane),  frag_h16(bs, 0, lane),  acc);
    acc = wmma_f16(frag_f32(a3, 32, lane), frag_h16(bs, 32, lane), acc);
#pragma unroll
    for (int rr = 0; rr < 8; ++rr) {
      int o = mt * 16 + rr + 8 * half, p = nt * 16 + lm;
      h[((size_t)(b * WCH + o) << 11) + s0 + p] = gelu_exact(acc[rr] + b2[o] + wb[o]);
    }
  }
}

// ---------------------------------------------------------------------------
// 8. Decoder: out = W2 @ gelu(W1 @ h + b1) + b2    (64 -> 128 -> 8)
// ---------------------------------------------------------------------------
#define TPAD2 136

__global__ void lno_decoder_kernel(const float* __restrict__ h,
                                   const float* __restrict__ w1, const float* __restrict__ b1,
                                   const float* __restrict__ w2, const float* __restrict__ b2,
                                   float* __restrict__ out) {
  int b  = blockIdx.x >> 5;
  int s0 = (blockIdx.x & 31) << 6;
  __shared__ _Float16 TH[64 * TPAD];    // [pos][64ch]
  __shared__ _Float16 T1[64 * TPAD2];   // [pos][128ch]
  int t = threadIdx.x;
  int ch = t >> 2, pos0 = (t & 3) << 4;
  const float* sh = h + ((size_t)(b * WCH + ch) << 11) + s0 + pos0;
#pragma unroll
  for (int i = 0; i < 16; ++i)
    TH[(pos0 + i) * TPAD + ch] = (_Float16)sh[i];
  __syncthreads();

  int lane = t & 31, wv = t >> 5;
  int half = lane >> 4, lm = lane & 15;

  // GEMM1: 128x64pos, K=64: 8 mtiles x 4 ntiles = 32 tiles, 4 per wave.
#pragma unroll
  for (int tt = 0; tt < 4; ++tt) {
    int id = wv * 4 + tt, mt = id >> 2, nt = id & 3;
    const float*    ar = w1 + (mt * 16 + lm) * WCH;
    const _Float16* br = &TH[(nt * 16 + lm) * TPAD];
    v8f acc = {};
    acc = wmma_f16(frag_f32(ar, 0, lane),  frag_h16(br, 0, lane),  acc);
    acc = wmma_f16(frag_f32(ar, 32, lane), frag_h16(br, 32, lane), acc);
#pragma unroll
    for (int rr = 0; rr < 8; ++rr) {
      int o = mt * 16 + rr + 8 * half, p = nt * 16 + lm;
      T1[p * TPAD2 + o] = (_Float16)gelu_exact(acc[rr] + b1[o]);
    }
  }
  __syncthreads();

  // GEMM2: 8x64pos, K=128. Pad M 8->16 with zeroed A rows; waves 0..3 only.
  if (wv < 4) {
    int nt = wv;
    const _Float16* br = &T1[(nt * 16 + lm) * TPAD2];
    v8f acc = {};
    for (int k0 = 0; k0 < 128; k0 += 32) {
      v16h a = {};
      if (lm < 8) a = frag_f32(w2 + lm * 128, k0, lane);
      acc = wmma_f16(a, frag_h16(br, k0, lane), acc);
    }
#pragma unroll
    for (int rr = 0; rr < 8; ++rr) {
      int o = rr + 8 * half;
      if (o < 8) {
        int p = nt * 16 + lm;
        out[((size_t)(b * 8 + o) << 11) + s0 + p] = acc[rr] + b2[o];
      }
    }
  }
}

// ---------------------------------------------------------------------------
// Host orchestration
// ---------------------------------------------------------------------------
extern "C" void kernel_launch(void* const* d_in, const int* in_sizes, int n_in,
                              void* d_out, int out_size, void* d_ws, size_t ws_size,
                              hipStream_t stream) {
  (void)in_sizes; (void)n_in; (void)out_size; (void)ws_size;
  const float* x      = (const float*)d_in[0];
  const float* enc_w  = (const float*)d_in[1];
  const float* enc_b  = (const float*)d_in[2];
  const float* lconv  = (const float*)d_in[3];   // [4][64][64][64]
  const float* w_w    = (const float*)d_in[4];   // [4][64][64]
  const float* w_b    = (const float*)d_in[5];   // [4][64]
  const float* f_w1   = (const float*)d_in[6];
  const float* f_b1   = (const float*)d_in[7];
  const float* f_w2   = (const float*)d_in[8];
  const float* f_b2   = (const float*)d_in[9];
  const float* dec_w1 = (const float*)d_in[10];  // [128][64]
  const float* dec_b1 = (const float*)d_in[11];
  const float* dec_w2 = (const float*)d_in[12];  // [8][128]
  const float* dec_b2 = (const float*)d_in[13];
  float* outp = (float*)d_out;

  char* ws = (char*)d_ws;
  float*     h    = (float*)(ws);                                   // 16 MB
  float*     y    = (float*)(ws + (size_t)16 * 1024 * 1024);        // 16 MB
  _Float16*  nbuf = (_Float16*)(ws + (size_t)32 * 1024 * 1024);     //  8 MB (hn / yn shared)
  float*     xs   = (float*)(ws + (size_t)40 * 1024 * 1024);        // 512 KB
  _Float16*  spec = (_Float16*)(ws + (size_t)40 * 1024 * 1024 + 512 * 1024);
  _Float16*  Gms  = (_Float16*)(ws + (size_t)41 * 1024 * 1024);     // 256 KB
  _Float16*  Gsm  = (_Float16*)(ws + (size_t)41 * 1024 * 1024 + 256 * 1024);

  dim3 blk(256);

  lno_basis_kernel<<<512, blk, 0, stream>>>(Gms, Gsm);
  lno_encoder_kernel<<<16384, blk, 0, stream>>>(x, enc_w, enc_b, h);

  for (int i = 0; i < 4; ++i) {
    lno_inorm_kernel<<<2048, blk, 0, stream>>>(h, nbuf);
    lno_proj_gemm_kernel<<<64, blk, 0, stream>>>(nbuf, Gms, xs);
    lno_modemix_kernel<<<512, blk, 0, stream>>>(xs, lconv + (size_t)i * 64 * 64 * 64, spec);
    lno_recon_gemm_kernel<<<2048, blk, 0, stream>>>(spec, Gsm, y);
    lno_inorm_kernel<<<2048, blk, 0, stream>>>(y, nbuf);
    lno_ffn_kernel<<<1024, blk, 0, stream>>>(nbuf, h,
                                             f_w1 + (size_t)i * 4096, f_b1 + (size_t)i * 64,
                                             f_w2 + (size_t)i * 4096, f_b2 + (size_t)i * 64,
                                             w_w  + (size_t)i * 4096, w_b  + (size_t)i * 64);
  }

  lno_decoder_kernel<<<1024, blk, 0, stream>>>(h, dec_w1, dec_b1, dec_w2, dec_b2, outp);
}
